// SSM_57088705298537
// MI455X (gfx1250) — matrix operand order repaired and checked by reading, not compiled, not run on hardware
//
#include <hip/hip_runtime.h>

// ---------------- model dims ----------------
#define S_DIM   64
#define A_DIM   4
#define H_DIM   1024
#define HID     512
#define DEC_HID 1024
#define IMG     (3 * 64 * 64)           // 12288
#define BATCH   64
#define TSTEPS  30
#define QW      (S_DIM + H_DIM + A_DIM) // 1092
#define PW      (S_DIM + A_DIM)         // 68
#define QWP     1120                    // QW padded to mult of 32
#define PWP     96                      // PW padded to mult of 32
#define MIN_STD 0.01f
#define LOG2PI_F 1.8378770664093453f
#define GAMMA_F  1.0f

typedef __attribute__((ext_vector_type(16))) __bf16 v16bf;
typedef __attribute__((ext_vector_type(8)))  __bf16 v8bf;
typedef __attribute__((ext_vector_type(8)))  float  v8f;
typedef __attribute__((ext_vector_type(4)))  float  v4f;

__device__ __forceinline__ float softplusf(float x) {
  return fmaxf(x, 0.f) + log1pf(expf(-fabsf(x)));
}

// ---------------------------------------------------------------------------
// Weight prep: Wt[n][k] = bf16(W[k][n]), K zero-padded to Kp.
// One-shot per launch; reads stride-N, writes coalesced bf16.
// ---------------------------------------------------------------------------
__global__ __launch_bounds__(256)
void ssm_wprep(const float* __restrict__ W, __bf16* __restrict__ Wt,
               int K, int N, int Kp) {
  int idx = blockIdx.x * 256 + threadIdx.x;
  if (idx >= N * Kp) return;
  int n = idx / Kp, k = idx - n * Kp;
  float v = (k < K) ? W[(size_t)k * N + n] : 0.f;
  Wt[idx] = (__bf16)v;
}

// ---------------------------------------------------------------------------
// GEMM: C[M,N] = act(A[M,Kp] @ Wt^T + bias)
//   A : fp32 row-major, lda = Kp (zero-padded K), converted to bf16 in-reg
//   Wt: bf16, N-major [N][Kp]  (pre-transposed weights)
// No LDS, no bounds checks: M%64==0, N%(32*NSUB)==0, Kp%32==0 by caller.
// Block: 256 thr = 8 waves (4 M-subtiles x 2 N-waves); wave computes a
// 16 x (16*NSUB) strip: one A fragment, NSUB independent WMMAs per K-step.
// launch_bounds(256,1): allow full VGPR budget so all NSUB B-fragments stay
// live in distinct registers -> loads clause together, partial loadcnt waits.
// Fragment layouts per CDNA5 ISA 7.12.2:
//   A 16x32 bf16 : lane&15 = row; lanes 0-15 k-octets {0-7,16-23},
//                  lanes 16-31 {8-15,24-31}  -> 4x b128 fp32 loads + cvt
//   B 32x16 bf16 : lane&15 = col; lanes 0-15 k-octets {0-7,16-23}, lanes 16-31 {8-15,24-31}
//                  -> 2x b128 bf16 loads from N-major Wt
// ACT: 0 none, 1 relu, 2 tanh.  NSUB: 2 (BN=64) or 4 (BN=128).
// ---------------------------------------------------------------------------
template <int ACT, int NSUB>
__global__ __launch_bounds__(256, 1)
void ssm_gemm_wmma(const float* __restrict__ A, int lda,
                   const __bf16* __restrict__ Wt, int ldw,
                   const float* __restrict__ bias,
                   float* __restrict__ C,
                   int N, int Kp) {
  const int lane  = threadIdx.x & 31;
  const int wave  = threadIdx.x >> 5;
  const int wm    = wave & 3;
  const int wn    = wave >> 2;
  const int waveN = 16 * NSUB;
  const int m0    = blockIdx.x * 64;
  const int n0    = blockIdx.y * (2 * waveN);

  const int mA   = m0 + wm * 16 + (lane & 15);       // A fragment row
  const int koct = (lane >> 4) * 8;                  // A k-octet offset
  const int ksb  = (lane >> 4) * 8;                  // B k-octet offset
  const int nw   = n0 + wn * waveN + (lane & 15);    // B subtile-0 column

  const float* aRow = A + (size_t)mA * lda + koct;
  const __bf16* bRow[NSUB];
#pragma unroll
  for (int s = 0; s < NSUB; ++s)
    bRow[s] = Wt + (size_t)(nw + s * 16) * ldw + ksb;

  v8f acc[NSUB];
#pragma unroll
  for (int s = 0; s < NSUB; ++s) acc[s] = (v8f){};

  for (int kb = 0; kb < Kp; kb += 32) {
    // issue ALL fragment loads up front (distinct regs -> one clause,
    // partial loadcnt waits, loads overlap preceding WMMAs)
    v16bf bfv[NSUB];
#pragma unroll
    for (int s = 0; s < NSUB; ++s) {
      v8bf lo = *(const v8bf*)(bRow[s] + kb);
      v8bf hi = *(const v8bf*)(bRow[s] + kb + 16);
      bfv[s] = __builtin_shufflevector(lo, hi,
          0, 1, 2, 3, 4, 5, 6, 7, 8, 9, 10, 11, 12, 13, 14, 15);
    }
    v4f a0 = *(const v4f*)(aRow + kb);
    v4f a1 = *(const v4f*)(aRow + kb + 4);
    v4f a2 = *(const v4f*)(aRow + kb + 16);
    v4f a3 = *(const v4f*)(aRow + kb + 20);

    // speculative prefetch of next K-tile (global_prefetch_b8; OOB dropped)
    __builtin_prefetch(aRow + kb + 32, 0, 1);
#pragma unroll
    for (int s = 0; s < NSUB; ++s)
      __builtin_prefetch(bRow[s] + kb + 32, 0, 1);

    v16bf af;
#pragma unroll
    for (int i = 0; i < 4; ++i) {
      af[i]      = (__bf16)a0[i];
      af[4 + i]  = (__bf16)a1[i];
      af[8 + i]  = (__bf16)a2[i];
      af[12 + i] = (__bf16)a3[i];
    }
#pragma unroll
    for (int s = 0; s < NSUB; ++s) {
      acc[s] = __builtin_amdgcn_wmma_f32_16x16x32_bf16(
          false, af, false, bfv[s], (short)0, acc[s], false, false);
    }
  }

  // epilogue: C/D layout -> lane&15 = N col; rows r + 8*(lane>>4)
  const int mo = m0 + wm * 16 + ((lane >> 4) * 8);
#pragma unroll
  for (int s = 0; s < NSUB; ++s) {
    const int nc = nw + s * 16;
    const float bv = bias[nc];
#pragma unroll
    for (int r = 0; r < 8; ++r) {
      float v = acc[s][r] + bv;
      if (ACT == 1)      v = fmaxf(v, 0.f);
      else if (ACT == 2) v = tanhf(v);
      C[(size_t)(mo + r) * N + nc] = v;
    }
  }
}

// ---------------------------------------------------------------------------
// concat with zero padding: Qin[b] = [s|h|a|0...] (QWP), Pin[b] = [s|a|0...] (PWP)
// ---------------------------------------------------------------------------
__global__ void ssm_concat(const float* __restrict__ s, const float* __restrict__ h,
                           const float* __restrict__ a,
                           float* __restrict__ Qin, float* __restrict__ Pin) {
  int i = blockIdx.x * blockDim.x + threadIdx.x;
  const int nq = BATCH * QWP;
  const int np = BATCH * PWP;
  if (i < nq) {
    int b = i / QWP, j = i - b * QWP;
    float v = 0.f;
    if (j < S_DIM)              v = s ? s[b * S_DIM + j] : 0.f;
    else if (j < S_DIM + H_DIM) v = h[b * H_DIM + (j - S_DIM)];
    else if (j < QW)            v = a ? a[b * A_DIM + (j - S_DIM - H_DIM)] : 0.f;
    Qin[i] = v;
  } else if (i < nq + np) {
    int k = i - nq;
    int b = k / PWP, j = k - b * PWP;
    float v = 0.f;
    if (j < S_DIM)      v = s ? s[b * S_DIM + j] : 0.f;
    else if (j < PW)    v = a ? a[b * A_DIM + (j - S_DIM)] : 0.f;
    Pin[k] = v;
  }
}

// ---------------------------------------------------------------------------
// KL / aux losses + rsample; s_out = ql + (softplus(qs)+MIN_STD)*eps
// acc[0] += s_loss, acc[1] += s_aux
// ---------------------------------------------------------------------------
__global__ __launch_bounds__(256)
void ssm_kl_sample(const float* __restrict__ ql, const float* __restrict__ qs_pre,
                   const float* __restrict__ pl, const float* __restrict__ ps_pre,
                   const float* __restrict__ eps, float* __restrict__ s_out,
                   float* __restrict__ acc) {
  __shared__ float red0[256];
  __shared__ float red1[256];
  int i = blockIdx.x * 256 + threadIdx.x;
  float kl = 0.f, aux = 0.f;
  if (i < BATCH * S_DIM) {
    float qlv = ql[i], plv = pl[i];
    float qs = softplusf(qs_pre[i]) + MIN_STD;
    float ps = softplusf(ps_pre[i]) + MIN_STD;
    float d  = qlv - plv;
    kl  = logf(ps / qs) + (qs * qs + d * d) / (2.f * ps * ps) - 0.5f;
    aux = -logf(qs) + 0.5f * (qs * qs + qlv * qlv) - 0.5f;
    s_out[i] = qlv + qs * eps[i];
  }
  red0[threadIdx.x] = kl;
  red1[threadIdx.x] = aux;
  __syncthreads();
  for (int st = 128; st > 0; st >>= 1) {
    if (threadIdx.x < st) {
      red0[threadIdx.x] += red0[threadIdx.x + st];
      red1[threadIdx.x] += red1[threadIdx.x + st];
    }
    __syncthreads();
  }
  if (threadIdx.x == 0) {
    atomicAdd(&acc[0], red0[0] * (1.f / BATCH));
    atomicAdd(&acc[1], red1[0] * (1.f / (BATCH * S_DIM)));
  }
}

// ---------------------------------------------------------------------------
// image NLLs -> acc[2..5]
// ---------------------------------------------------------------------------
__global__ __launch_bounds__(256)
void ssm_img_loss(const float* __restrict__ xq, const float* __restrict__ xp,
                  const float* __restrict__ xt,
                  const float* __restrict__ xq_prev, const float* __restrict__ xp_prev,
                  const float* __restrict__ x_prev,
                  float* __restrict__ acc, int with_delta) {
  __shared__ float r0[256], r1[256], r2[256], r3[256];
  int i = blockIdx.x * 256 + threadIdx.x;
  float lq = 0.f, lp = 0.f, ldq = 0.f, ldp = 0.f;
  if (i < BATCH * IMG) {
    float x  = xt[i];
    float vq = xq[i];
    float vp = xp[i];
    float eq = x - vq, ep = x - vp;
    lq = 0.5f * eq * eq + 0.5f * LOG2PI_F;
    lp = 0.5f * ep * ep + 0.5f * LOG2PI_F;
    if (with_delta) {
      float dx = x - x_prev[i];
      float e1 = dx - (vq - xq_prev[i]);
      float e2 = dx - (vp - xp_prev[i]);
      ldq = 0.5f * e1 * e1 + 0.5f * LOG2PI_F;
      ldp = 0.5f * e2 * e2 + 0.5f * LOG2PI_F;
    }
  }
  r0[threadIdx.x] = lq;  r1[threadIdx.x] = lp;
  r2[threadIdx.x] = ldq; r3[threadIdx.x] = ldp;
  __syncthreads();
  for (int st = 128; st > 0; st >>= 1) {
    if (threadIdx.x < st) {
      r0[threadIdx.x] += r0[threadIdx.x + st];
      r1[threadIdx.x] += r1[threadIdx.x + st];
      r2[threadIdx.x] += r2[threadIdx.x + st];
      r3[threadIdx.x] += r3[threadIdx.x + st];
    }
    __syncthreads();
  }
  if (threadIdx.x == 0) {
    const float inv_b = 1.f / BATCH;
    atomicAdd(&acc[2], r0[0] * inv_b);
    atomicAdd(&acc[3], r1[0] * inv_b);
    if (with_delta) {
      atomicAdd(&acc[4], r2[0] * inv_b);
      atomicAdd(&acc[5], r3[0] * inv_b);
    }
  }
}

__global__ void ssm_zero_acc(float* acc) {
  if (threadIdx.x < 8) acc[threadIdx.x] = 0.f;
}

__global__ void ssm_final(const float* __restrict__ acc, float* __restrict__ out) {
  out[0] = acc[0] + acc[2] + acc[3] + acc[4] + acc[5] + GAMMA_F * acc[1];
}

// ---------------------------------------------------------------------------
extern "C" void kernel_launch(void* const* d_in, const int* in_sizes, int n_in,
                              void* d_out, int out_size, void* d_ws, size_t ws_size,
                              hipStream_t stream) {
  (void)in_sizes; (void)n_in; (void)out_size; (void)ws_size;
  const float* x0     = (const float*)d_in[0];
  const float* x      = (const float*)d_in[1];
  const float* a_in   = (const float*)d_in[2];
  const float* eps_in = (const float*)d_in[3];
  const float* We1 = (const float*)d_in[4];  const float* be1 = (const float*)d_in[5];
  const float* We2 = (const float*)d_in[6];  const float* be2 = (const float*)d_in[7];
  const float* Wp1 = (const float*)d_in[8];  const float* bp1 = (const float*)d_in[9];
  const float* Wp_loc = (const float*)d_in[10]; const float* bp_loc = (const float*)d_in[11];
  const float* Wp_sc  = (const float*)d_in[12]; const float* bp_sc  = (const float*)d_in[13];
  const float* Wq1 = (const float*)d_in[14]; const float* bq1 = (const float*)d_in[15];
  const float* Wq_loc = (const float*)d_in[16]; const float* bq_loc = (const float*)d_in[17];
  const float* Wq_sc  = (const float*)d_in[18]; const float* bq_sc  = (const float*)d_in[19];
  const float* Wd1 = (const float*)d_in[20]; const float* bd1 = (const float*)d_in[21];
  const float* Wd2 = (const float*)d_in[22]; const float* bd2 = (const float*)d_in[23];

  // ---- workspace layout (float units; every offset 64B-aligned) ----
  float* ws = (float*)d_ws;
  size_t off = 0;
  auto alloc_f = [&](size_t nfloats) {
    float* p = ws + off;
    off += (nfloats + 15) & ~(size_t)15;
    return p;
  };
  auto alloc_b = [&](size_t nbf16) {
    return (__bf16*)alloc_f((nbf16 + 1) / 2);
  };

  const size_t ROWS = (size_t)(TSTEPS + 1) * BATCH;  // 1984
  // bf16 transposed weights
  __bf16* We1t = alloc_b((size_t)1024 * IMG);
  __bf16* We2t = alloc_b((size_t)H_DIM * 1024);
  __bf16* Wq1t = alloc_b((size_t)HID * QWP);
  __bf16* Wp1t = alloc_b((size_t)HID * PWP);
  __bf16* Wqlt = alloc_b((size_t)S_DIM * HID);
  __bf16* Wqst = alloc_b((size_t)S_DIM * HID);
  __bf16* Wplt = alloc_b((size_t)S_DIM * HID);
  __bf16* Wpst = alloc_b((size_t)S_DIM * HID);
  __bf16* Wd1t = alloc_b((size_t)DEC_HID * S_DIM);
  __bf16* Wd2t = alloc_b((size_t)IMG * DEC_HID);
  // fp32 activations
  float* e1    = alloc_f(ROWS * 1024);
  float* h_all = alloc_f(ROWS * H_DIM);
  float* Qin   = alloc_f((size_t)BATCH * QWP);
  float* Pin   = alloc_f((size_t)BATCH * PWP);
  float* Hq    = alloc_f((size_t)BATCH * HID);
  float* Hp    = alloc_f((size_t)BATCH * HID);
  float* ql_   = alloc_f((size_t)BATCH * S_DIM);
  float* qs_   = alloc_f((size_t)BATCH * S_DIM);
  float* pl_   = alloc_f((size_t)BATCH * S_DIM);
  float* ps_   = alloc_f((size_t)BATCH * S_DIM);
  float* s_buf = alloc_f((size_t)BATCH * S_DIM);
  float* Hd    = alloc_f((size_t)BATCH * DEC_HID);
  float* xqb[2]; float* xpb[2];
  xqb[0] = alloc_f((size_t)BATCH * IMG);
  xqb[1] = alloc_f((size_t)BATCH * IMG);
  xpb[0] = alloc_f((size_t)BATCH * IMG);
  xpb[1] = alloc_f((size_t)BATCH * IMG);
  float* acc = alloc_f(8);

  const dim3 blk(256);
  auto wgrid = [](size_t n) { return dim3((unsigned)((n + 255) / 256)); };

  ssm_zero_acc<<<1, 8, 0, stream>>>(acc);

  // ---- one-shot weight transpose + bf16 convert (+K zero-pad) ----
  ssm_wprep<<<wgrid((size_t)1024 * IMG), blk, 0, stream>>>(We1, We1t, IMG, 1024, IMG);
  ssm_wprep<<<wgrid((size_t)H_DIM * 1024), blk, 0, stream>>>(We2, We2t, 1024, H_DIM, 1024);
  ssm_wprep<<<wgrid((size_t)HID * QWP), blk, 0, stream>>>(Wq1, Wq1t, QW, HID, QWP);
  ssm_wprep<<<wgrid((size_t)HID * PWP), blk, 0, stream>>>(Wp1, Wp1t, PW, HID, PWP);
  ssm_wprep<<<wgrid((size_t)S_DIM * HID), blk, 0, stream>>>(Wq_loc, Wqlt, HID, S_DIM, HID);
  ssm_wprep<<<wgrid((size_t)S_DIM * HID), blk, 0, stream>>>(Wq_sc, Wqst, HID, S_DIM, HID);
  ssm_wprep<<<wgrid((size_t)S_DIM * HID), blk, 0, stream>>>(Wp_loc, Wplt, HID, S_DIM, HID);
  ssm_wprep<<<wgrid((size_t)S_DIM * HID), blk, 0, stream>>>(Wp_sc, Wpst, HID, S_DIM, HID);
  ssm_wprep<<<wgrid((size_t)DEC_HID * S_DIM), blk, 0, stream>>>(Wd1, Wd1t, S_DIM, DEC_HID, S_DIM);
  ssm_wprep<<<wgrid((size_t)IMG * DEC_HID), blk, 0, stream>>>(Wd2, Wd2t, DEC_HID, IMG, DEC_HID);

  // ---- encoder (parallel over all 31 frames); N=1024 -> NSUB=4 (BN=128) ----
  ssm_gemm_wmma<1, 4><<<dim3(1, 1024 / 128), blk, 0, stream>>>(
      x0, IMG, We1t, IMG, be1, e1, 1024, IMG);
  ssm_gemm_wmma<1, 4><<<dim3(TSTEPS * BATCH / 64, 1024 / 128), blk, 0, stream>>>(
      x, IMG, We1t, IMG, be1, e1 + (size_t)BATCH * 1024, 1024, IMG);
  ssm_gemm_wmma<0, 4><<<dim3((unsigned)(ROWS / 64), H_DIM / 128), blk, 0, stream>>>(
      e1, 1024, We2t, 1024, be2, h_all, H_DIM, 1024);

  const unsigned concat_blocks = (BATCH * (QWP + PWP) + 255) / 256;

  // ---- s0 = posterior mean with s_prev = 0, a = 0, h = h0 ----
  ssm_concat<<<concat_blocks, blk, 0, stream>>>(nullptr, h_all, nullptr, Qin, Pin);
  ssm_gemm_wmma<1, 4><<<dim3(1, HID / 128), blk, 0, stream>>>(
      Qin, QWP, Wq1t, QWP, bq1, Hq, HID, QWP);
  ssm_gemm_wmma<0, 2><<<dim3(1, 1), blk, 0, stream>>>(
      Hq, HID, Wqlt, HID, bq_loc, s_buf, S_DIM, HID);

  // ---- sequential scan over T steps ----
  for (int t = 0; t < TSTEPS; ++t) {
    const float* h_t = h_all + (size_t)(t + 1) * BATCH * H_DIM;
    const float* a_t = a_in + (size_t)t * BATCH * A_DIM;
    const float* e_t = eps_in + (size_t)t * BATCH * S_DIM;
    const float* x_t = x + (size_t)t * BATCH * IMG;

    ssm_concat<<<concat_blocks, blk, 0, stream>>>(s_buf, h_t, a_t, Qin, Pin);
    ssm_gemm_wmma<1, 4><<<dim3(1, HID / 128), blk, 0, stream>>>(
        Qin, QWP, Wq1t, QWP, bq1, Hq, HID, QWP);
    ssm_gemm_wmma<1, 4><<<dim3(1, HID / 128), blk, 0, stream>>>(
        Pin, PWP, Wp1t, PWP, bp1, Hp, HID, PWP);
    ssm_gemm_wmma<0, 2><<<dim3(1, 1), blk, 0, stream>>>(
        Hq, HID, Wqlt, HID, bq_loc, ql_, S_DIM, HID);
    ssm_gemm_wmma<0, 2><<<dim3(1, 1), blk, 0, stream>>>(
        Hq, HID, Wqst, HID, bq_sc, qs_, S_DIM, HID);
    ssm_gemm_wmma<0, 2><<<dim3(1, 1), blk, 0, stream>>>(
        Hp, HID, Wplt, HID, bp_loc, pl_, S_DIM, HID);
    ssm_gemm_wmma<0, 2><<<dim3(1, 1), blk, 0, stream>>>(
        Hp, HID, Wpst, HID, bp_sc, ps_, S_DIM, HID);

    // KL + aux losses, rsample -> s_buf (carry for next step)
    ssm_kl_sample<<<wgrid(BATCH * S_DIM), blk, 0, stream>>>(
        ql_, qs_, pl_, ps_, e_t, s_buf, acc);

    const int cur = t & 1, prv = cur ^ 1;
    // decoder on sampled posterior state
    ssm_gemm_wmma<1, 4><<<dim3(1, DEC_HID / 128), blk, 0, stream>>>(
        s_buf, S_DIM, Wd1t, S_DIM, bd1, Hd, DEC_HID, S_DIM);
    ssm_gemm_wmma<2, 4><<<dim3(1, IMG / 128), blk, 0, stream>>>(
        Hd, DEC_HID, Wd2t, DEC_HID, bd2, xqb[cur], IMG, DEC_HID);
    // decoder on prior mean
    ssm_gemm_wmma<1, 4><<<dim3(1, DEC_HID / 128), blk, 0, stream>>>(
        pl_, S_DIM, Wd1t, S_DIM, bd1, Hd, DEC_HID, S_DIM);
    ssm_gemm_wmma<2, 4><<<dim3(1, IMG / 128), blk, 0, stream>>>(
        Hd, DEC_HID, Wd2t, DEC_HID, bd2, xpb[cur], IMG, DEC_HID);

    const float* x_prev = (t == 0) ? nullptr : x + (size_t)(t - 1) * BATCH * IMG;
    ssm_img_loss<<<wgrid((size_t)BATCH * IMG), blk, 0, stream>>>(
        xqb[cur], xpb[cur], x_t, xqb[prv], xpb[prv], x_prev, acc, t > 0 ? 1 : 0);
  }

  ssm_final<<<1, 1, 0, stream>>>(acc, (float*)d_out);
}
